// CombinedSegmentationLoss_41446434406854
// MI455X (gfx1250) — compile-verified, matrix operand order checked by reading, not run on hardware
//
#include <hip/hip_runtime.h>
#include <hip/hip_bf16.h>
#include <math.h>

// ---------------------------------------------------------------------------
// CombinedSegmentationLoss for MI455X (gfx1250, wave32)
//  - sort-free OHEM + Lovasz via per-image histograms (LDS-privatized)
//  - Tversky reductions (tp, sum_p, sum_t) on the matrix pipe via
//    v_wmma_f32_16x16x32_f16 (diagonal-dot trick, f32 accumulate)
//  - logits staged into LDS with the Tensor Data Mover (tensor_load_to_lds),
//    double-buffered, synchronized with s_wait_tensorcnt
// ---------------------------------------------------------------------------

typedef __attribute__((ext_vector_type(16))) _Float16 v16h;
typedef __attribute__((ext_vector_type(8)))  float    v8f;
typedef __attribute__((ext_vector_type(4)))  unsigned v4u;
typedef __attribute__((ext_vector_type(8)))  unsigned v8u;

#define NBINS     1024
#define BINSCALE  (NBINS / 16.0f)   // bins span 16.0 units of value range
#define SCW       16
#define IMG_WORDS (5 * NBINS + SCW)
#define BATCH     16
#define TPB       256
#define CHUNK     (TPB * 16)        // 4096 elements staged per step

// scalar slot indices (per image, after the 5 histogram arrays)
// 0:tp 1:sumP 2:sumT 3:posSum 4:nPos 5:maxBits 6:hasTissue 7:loss0
// 8:ohem_i 9:ft_i 10:lov_i 11:posb

__device__ __forceinline__ float wave_sum32(float v) {
#pragma unroll
  for (int off = 16; off > 0; off >>= 1) v += __shfl_xor(v, off, 32);
  return v;
}
__device__ __forceinline__ float wave_max32(float v) {
#pragma unroll
  for (int off = 16; off > 0; off >>= 1) v = fmaxf(v, __shfl_xor(v, off, 32));
  return v;
}

// Issue one TDM transfer: copy a 1-row tile of `tileElems` f32 from global
// memory into LDS at byte offset ldsOff. Elements past `nElems` zero-fill
// (OOB reads return zero per the D# tensor_dim0 bound).
__device__ __forceinline__ void tdm_load_row_f32(unsigned ldsOff,
                                                 const void* gptr,
                                                 unsigned nElems,
                                                 unsigned tileElems) {
  unsigned long long ga = (unsigned long long)(uintptr_t)gptr;
  unsigned long long stride0 = nElems ? nElems : 1u;
  v4u g0;
  g0[0] = 1u;                                            // count=1 (valid user D#)
  g0[1] = ldsOff;                                        // lds_addr (bytes)
  g0[2] = (unsigned)(ga & 0xFFFFFFFFu);                  // global_addr[31:0]
  g0[3] = (unsigned)((ga >> 32) & 0x01FFFFFFu)           // global_addr[56:32]
          | (2u << 30);                                  // type=2 ("image")
  v8u g1;
  g1[0] = (2u << 16);                                    // data_size=2 -> 4 bytes
  g1[1] = (nElems & 0xFFFFu) << 16;                      // tensor_dim0[15:0]
  g1[2] = (nElems >> 16) | (1u << 16);                   // tensor_dim0[31:16] | tensor_dim1=1
  g1[3] = (tileElems & 0xFFFFu) << 16;                   // tile_dim0
  g1[4] = 1u;                                            // tile_dim1=1, tile_dim2=0
  g1[5] = (unsigned)(stride0 & 0xFFFFFFFFu);             // tensor_dim0_stride lo
  g1[6] = (unsigned)((stride0 >> 32) & 0xFFFFu);         // tensor_dim0_stride hi
  g1[7] = 0u;
  asm volatile("tensor_load_to_lds %0, %1" :: "s"(g0), "s"(g1) : "memory");
}

__global__ void seg_loss_zero(unsigned* __restrict__ ws, long n) {
  long stride = (long)gridDim.x * blockDim.x;
  for (long i = (long)blockIdx.x * blockDim.x + threadIdx.x; i < n; i += stride)
    ws[i] = 0u;
}

__global__ __launch_bounds__(TPB) void seg_loss_pass_a(
    const float* __restrict__ logits, const int* __restrict__ targets,
    const float* __restrict__ mask, float* __restrict__ ws, int P) {
  __shared__ unsigned sNegCnt[NBINS];
  __shared__ float    sNegSum[NBINS];
  __shared__ unsigned sErrCnt[NBINS];
  __shared__ unsigned sErrPos[NBINS];
  __shared__ float    sErrRelu[NBINS];
  __shared__ float    sL[2][CHUNK];       // TDM-staged logits (double buffer)
  __shared__ float    sAcc[5];            // tp, sumP, sumT, posSum, nPos
  __shared__ int      sMaxBits;
  __shared__ unsigned sHas;

  const int tid = threadIdx.x;
  for (int b = tid; b < NBINS; b += TPB) {
    sNegCnt[b] = 0u; sNegSum[b] = 0.f;
    sErrCnt[b] = 0u; sErrPos[b] = 0u; sErrRelu[b] = 0.f;
  }
  if (tid == 0) {
    sAcc[0] = sAcc[1] = sAcc[2] = sAcc[3] = sAcc[4] = 0.f;
    sMaxBits = 0; sHas = 0u;
  }

  const int img = blockIdx.y;
  const size_t ibase = (size_t)img * (size_t)P;
  const float* lp = logits + ibase;
  const int*   tg = targets + ibase;
  const float* mp = mask + ibase;

  v16h ones;
#pragma unroll
  for (int j = 0; j < 16; ++j) ones[j] = (_Float16)1.0f;
  v8f c_pt = {};  // accumulates sum(p*t) on diagonal
  v8f c_p1 = {};  // accumulates sum(p) in column 0 (rows 0..15)
  v8f c_1t = {};  // accumulates sum(t) in row 0 (cols 0..15)

  float posSum = 0.f, nPos = 0.f, lmax = 0.f;
  unsigned has = 0u;

  const int gstride = gridDim.x * CHUNK;
  const int firstBase = blockIdx.x * CHUNK;
  int cur = 0;

  // prologue: kick off first TDM stage (one issue per block, via wave 0)
  if (tid < 32 && firstBase < P) {
    tdm_load_row_f32((unsigned)(uintptr_t)(void*)&sL[0][0], lp + firstBase,
                     (unsigned)(P - firstBase), CHUNK);
  }
  __syncthreads();  // also covers the LDS histogram zero-init

  for (int base = firstBase; base < P; base += gstride) {
    int nextBase = base + gstride;
    if (tid < 32) {
      if (nextBase < P) {
        tdm_load_row_f32((unsigned)(uintptr_t)(void*)&sL[cur ^ 1][0],
                         lp + nextBase, (unsigned)(P - nextBase), CHUNK);
        __builtin_amdgcn_s_wait_tensorcnt(1);  // buffer `cur` complete
      } else {
        __builtin_amdgcn_s_wait_tensorcnt(0);
      }
    }
    __syncthreads();  // staged data visible to all waves

    if (nextBase + tid < P) {  // hint next chunk of t/m (global_prefetch_b8)
      __builtin_prefetch(tg + nextBase + tid, 0, 0);
      __builtin_prefetch(mp + nextBase + tid, 0, 0);
    }

    const float* lbuf = sL[cur];
    v16h pa, tb;
#pragma unroll
    for (int j = 0; j < 16; ++j) {
      int e = j * TPB + tid;
      int k = base + e;
      float pv = 0.f, tv = 0.f;
      if (k < P) {
        float x = lbuf[e];                    // from TDM-staged LDS
        float t = (float)tg[k];
        float m = mp[k];
        // fast-math BCE: v_exp_f32 / v_log_f32 instead of libm expansion
        float bce = fmaxf(x, 0.f) - x * t + __logf(1.f + __expf(-fabsf(x)));
        bce *= m;
        float posf = t * m;
        posSum += bce * posf;
        nPos += posf;
        if (m > 0.5f) { has = 1u; lmax = fmaxf(lmax, bce); }
        if (t == 0.f && m > 0.5f) {
          int nbn = (int)(bce * BINSCALE);
          nbn = nbn < 0 ? 0 : (nbn > NBINS - 1 ? NBINS - 1 : nbn);
          atomicAdd(&sNegCnt[nbn], 1u);
          atomicAdd(&sNegSum[nbn], bce);
        }
        float sgn = 2.f * t - 1.f;
        float err = 1.f - x * sgn;
        int eb = (int)((err + 8.f) * BINSCALE);
        eb = eb < 0 ? 0 : (eb > NBINS - 1 ? NBINS - 1 : eb);
        atomicAdd(&sErrCnt[eb], 1u);
        if (t != 0.f) atomicAdd(&sErrPos[eb], 1u);
        atomicAdd(&sErrRelu[eb], fmaxf(err, 0.f));
        if (k == 0) {  // per-image fallback pixel
          float* sc0 = ws + (size_t)img * IMG_WORDS + 5 * NBINS;
          sc0[7] = bce;
        }
        pv = 1.f / (1.f + __expf(-x));  // sigmoid prob
        tv = t;
      }
      pa[j] = (_Float16)pv;
      tb[j] = (_Float16)tv;
    }
    // matrix-pipe reductions (EXEC all ones here; wave-uniform control flow)
    c_pt = __builtin_amdgcn_wmma_f32_16x16x32_f16(false, pa, false, tb,
                                                  (short)0, c_pt, false, false);
    c_p1 = __builtin_amdgcn_wmma_f32_16x16x32_f16(false, pa, false, ones,
                                                  (short)0, c_p1, false, false);
    c_1t = __builtin_amdgcn_wmma_f32_16x16x32_f16(false, ones, false, tb,
                                                  (short)0, c_1t, false, false);

    __syncthreads();  // all waves done reading sL[cur] before it is restaged
    cur ^= 1;
  }

  // extract: diag(c_pt) = tp; col0 of c_p1 = sumP; row0 of c_1t = sumT
  const unsigned lane = (unsigned)(tid & 31);
  float tpc = 0.f, pc = 0.f, tc = 0.f;
#pragma unroll
  for (int r = 0; r < 8; ++r) {
    tpc += (lane == (unsigned)r || lane == (unsigned)(16 + r)) ? c_pt[r] : 0.f;
    pc  += (lane == 0u || lane == 16u) ? c_p1[r] : 0.f;
  }
  tc = (lane < 16u) ? c_1t[0] : 0.f;

  tpc    = wave_sum32(tpc);
  pc     = wave_sum32(pc);
  tc     = wave_sum32(tc);
  posSum = wave_sum32(posSum);
  nPos   = wave_sum32(nPos);
  lmax   = wave_max32(lmax);
  float hasf = wave_max32((float)has);
  if (lane == 0u) {
    atomicAdd(&sAcc[0], tpc);
    atomicAdd(&sAcc[1], pc);
    atomicAdd(&sAcc[2], tc);
    atomicAdd(&sAcc[3], posSum);
    atomicAdd(&sAcc[4], nPos);
    atomicMax(&sMaxBits, __float_as_int(lmax));
    if (hasf > 0.f) atomicOr(&sHas, 1u);
  }
  __syncthreads();

  // flush LDS histograms to the per-image global histograms
  float* F = ws + (size_t)img * IMG_WORDS;
  unsigned* U = (unsigned*)F;
  for (int b = tid; b < NBINS; b += TPB) {
    if (sNegCnt[b]) {
      atomicAdd(&U[b], sNegCnt[b]);
      atomicAdd(&F[NBINS + b], sNegSum[b]);
    }
    if (sErrCnt[b]) {
      atomicAdd(&U[2 * NBINS + b], sErrCnt[b]);
      atomicAdd(&F[4 * NBINS + b], sErrRelu[b]);
    }
    if (sErrPos[b]) atomicAdd(&U[3 * NBINS + b], sErrPos[b]);
  }
  if (tid == 0) {
    float* sc = F + 5 * NBINS;
    unsigned* scU = (unsigned*)sc;
    atomicAdd(&sc[0], sAcc[0]);
    atomicAdd(&sc[1], sAcc[1]);
    atomicAdd(&sc[2], sAcc[2]);
    atomicAdd(&sc[3], sAcc[3]);
    atomicAdd(&sc[4], sAcc[4]);
    atomicMax((int*)&scU[5], sMaxBits);
    atomicOr(&scU[6], sHas);
  }
}

__global__ __launch_bounds__(TPB) void seg_loss_pass_b(float* __restrict__ ws,
                                                       int P, int k_all) {
  __shared__ unsigned sNegCnt[NBINS];
  __shared__ float    sNegSum[NBINS];
  __shared__ unsigned sErrCnt[NBINS];
  __shared__ unsigned sErrPos[NBINS];
  __shared__ float    sErrRelu[NBINS];

  const int img = blockIdx.x;
  const int tid = threadIdx.x;
  float* F = ws + (size_t)img * IMG_WORDS;
  unsigned* U = (unsigned*)F;
  for (int b = tid; b < NBINS; b += TPB) {
    sNegCnt[b]  = U[b];
    sNegSum[b]  = F[NBINS + b];
    sErrCnt[b]  = U[2 * NBINS + b];
    sErrPos[b]  = U[3 * NBINS + b];
    sErrRelu[b] = F[4 * NBINS + b];
  }
  __syncthreads();

  if (tid == 0) {
    float* sc = F + 5 * NBINS;
    unsigned* scU = (unsigned*)sc;
    float tp = sc[0], sumP = sc[1], sumT = sc[2];
    float posSum = sc[3], nPosF = sc[4];
    float lmax = __int_as_float((int)scU[5]);
    unsigned has = scU[6];
    float loss0 = sc[7];
    (void)sumT;

    long totPos = 0, totE = 0;
    for (int b = 0; b < NBINS; ++b) {
      totPos += (long)sErrPos[b];
      totE   += (long)sErrCnt[b];
    }

    // ---- OHEM: fractional top-k over negative-loss histogram ----
    long nPos = (long)(nPosF + 0.5f);
    long n_remain = (long)k_all - nPos;
    if (n_remain < 0) n_remain = 0;
    long rem = n_remain;
    float negKept = 0.f;
    for (int b = NBINS - 1; b >= 0 && rem > 0; --b) {
      long c = (long)sNegCnt[b];
      if (!c) continue;
      if (c <= rem) { negKept += sNegSum[b]; rem -= c; }
      else          { negKept += sNegSum[b] * ((float)rem / (float)c); rem = 0; }
    }
    long nKept = n_remain - rem;
    long cnt = nPos + nKept;
    float ohem = (cnt > 0) ? (posSum + negKept) / (float)cnt
                           : (has ? lmax : loss0);

    // ---- focal Tversky ----
    float sumTf = (float)totPos;  // exact count of positives
    float fn = fmaxf(sumTf - tp, 0.f);
    float fp = fmaxf(sumP - tp, 0.f);
    float tv = (tp + 1e-6f) / (tp + 0.3f * fn + 0.7f * fp + 1e-6f);
    float ft = powf(fmaxf(1.f - tv, 0.f), 1.33f);

    // ---- Lovasz hinge via telescoped histogram walk ----
    float p = (float)totPos;
    float lov = 0.f;
    bool haveNeg = (totE - totPos) > 0;
    if (totPos > 0) {
      float cumPos = 0.f, cumNeg = 0.f, Jprev = 0.f;
      for (int b = NBINS - 1; b >= 0; --b) {
        unsigned c = sErrCnt[b];
        if (!c) continue;
        unsigned cp = sErrPos[b];
        cumPos += (float)cp;
        cumNeg += (float)(c - cp);
        float J = 1.f - (p - cumPos) / (p + cumNeg);
        if (haveNeg) lov += (sErrRelu[b] / (float)c) * (J - Jprev);
        else         lov += sErrRelu[b] * 0.5f * (J + Jprev);  // grad==jaccard case
        Jprev = J;
      }
    }

    sc[8]  = ohem;
    sc[9]  = ft;
    sc[10] = lov;
    sc[11] = (totPos > 0) ? 1.f : 0.f;
  }
}

__global__ void seg_loss_final(const float* __restrict__ ws,
                               float* __restrict__ out) {
  if (blockIdx.x == 0 && threadIdx.x == 0) {
    float os = 0.f, fs = 0.f, ls = 0.f, np = 0.f;
    for (int i = 0; i < BATCH; ++i) {
      const float* sc = ws + (size_t)i * IMG_WORDS + 5 * NBINS;
      os += sc[8];
      if (sc[11] > 0.5f) { fs += sc[9]; ls += sc[10]; np += 1.f; }
    }
    float r = os / (float)BATCH;
    if (np > 0.f) r += fs / np + 0.2f * (ls / np);
    out[0] = r;
  }
}

extern "C" void kernel_launch(void* const* d_in, const int* in_sizes, int n_in,
                              void* d_out, int out_size, void* d_ws,
                              size_t ws_size, hipStream_t stream) {
  const float* logits  = (const float*)d_in[0];
  const int*   targets = (const int*)d_in[1];
  const float* mask    = (const float*)d_in[2];
  float* out = (float*)d_out;
  float* ws  = (float*)d_ws;

  int total = in_sizes[0];
  int P = total / BATCH;                         // per-image pixel count
  int k_all = (int)((double)P * 0.3);
  if (k_all < 1) k_all = 1;

  long words = (long)BATCH * IMG_WORDS;
  int zb = (int)((words + TPB - 1) / TPB);
  seg_loss_zero<<<zb, TPB, 0, stream>>>((unsigned*)d_ws, words);

  dim3 gA(96, BATCH);
  seg_loss_pass_a<<<gA, TPB, 0, stream>>>(logits, targets, mask, ws, P);
  seg_loss_pass_b<<<BATCH, TPB, 0, stream>>>(ws, P, k_all);
  seg_loss_final<<<1, 32, 0, stream>>>(ws, out);
}
